// DKBATNet_2456721293923
// MI455X (gfx1250) — matrix operand order repaired and checked by reading, not compiled, not run on hardware
//
#include <hip/hip_runtime.h>
#include <hip/hip_bf16.h>
#include <math.h>

// ---------------- problem constants ----------------
#define N_NODES 14541
#define M_REL   237
#define E_EDGES 120000
#define XS_     100
#define OUT_    100          // HID == OUT == 100
#define H_      2
#define NOUT    200          // H * OUT
#define NPAD    208          // 200 padded to 13 WMMA N-tiles
#define NT_     13           // NPAD/16
#define K1      300
#define K1P     320
#define K2      500
#define K2P     512
#define ETILES  (E_EDGES / 16)   // 7500, exact

typedef __attribute__((ext_vector_type(16))) __bf16 v16bf;
typedef __attribute__((ext_vector_type(8)))  float  v8f;

__device__ __forceinline__ __bf16 f2bf(float f) {
  union { float f; unsigned u; } v; v.f = f;
  unsigned r = (v.u + 0x7FFFu + ((v.u >> 16) & 1u)) >> 16;  // RNE truncate
  unsigned short us = (unsigned short)r;
  __bf16 b;
  __builtin_memcpy(&b, &us, sizeof(us));
  return b;
}

__device__ __forceinline__ float eluf(float x) {
  return x > 0.f ? x : (__expf(x) - 1.f);
}

// ---------------- prep kernels ----------------
__global__ void k_l2norm_rows(const float* __restrict__ x, float* __restrict__ xn, int n) {
  int i = blockIdx.x * blockDim.x + threadIdx.x;
  if (i >= n) return;
  const float* xr = x + (size_t)i * XS_;
  float s = 0.f;
  for (int k = 0; k < XS_; ++k) s += xr[k] * xr[k];
  float inv = 1.f / fmaxf(sqrtf(s), 1e-12f);
  float* o = xn + (size_t)i * XS_;
  for (int k = 0; k < XS_; ++k) o[k] = xr[k] * inv;
}

// W [200 x Kreal] f32 -> Wfrag[kt][nt][lane][16] bf16 (WMMA B-operand order;
// kt-major so one k-step's 13 fragments are contiguous -> immediate offsets)
// B 16-bit 32x16 lane layout: n = nt*16 + (lane&15), K = kt*32 + ((j<8)?j:j+8) + 8*(lane>>4)
__global__ void k_prep_wfrag(const float* __restrict__ W, __bf16* __restrict__ Wf,
                             int Kreal, int KT) {
  int idx = blockIdx.x * blockDim.x + threadIdx.x;
  int total = KT * NT_ * 512;            // 512 = 32 lanes * 16 elems
  if (idx >= total) return;
  int j    = idx & 15;
  int lane = (idx >> 4) & 31;
  int nt   = (idx >> 9) % NT_;
  int kt   = idx / (NT_ * 512);
  int n = nt * 16 + (lane & 15);
  int k = kt * 32 + ((j < 8) ? j : (j + 8)) + 8 * (lane >> 4);
  float v = (n < NOUT && k < Kreal) ? W[(size_t)n * Kreal + k] : 0.f;
  Wf[idx] = f2bf(v);
}

// Gather edge features concat(feat[row], feat[col], g[etype]) into
// Afrag[etile][kt][lane][16] bf16 (WMMA A-operand order).
// A 16-bit 16x32 lane layout: edge = etile*16 + (lane&15),
//                             K = kt*32 + ((j<8)?j:j+8) + 8*(lane>>4)
__global__ void k_gather_A(const float* __restrict__ feat, int fdim,
                           const float* __restrict__ grel,
                           const int* __restrict__ row, const int* __restrict__ col,
                           const int* __restrict__ etype,
                           __bf16* __restrict__ Af, int Kreal, int KT) {
  int idx = blockIdx.x * blockDim.x + threadIdx.x;
  int total = ETILES * KT * 512;
  if (idx >= total) return;
  int j     = idx & 15;
  int lane  = (idx >> 4) & 31;
  int kt    = (idx >> 9) % KT;
  int etile = idx / (KT * 512);
  int e  = etile * 16 + (lane & 15);
  int k  = kt * 32 + ((j < 8) ? j : (j + 8)) + 8 * (lane >> 4);
  int fdim2 = fdim * 2;
  float v;
  if (k < fdim)       v = feat[(size_t)row[e] * fdim + k];
  else if (k < fdim2) v = feat[(size_t)col[e] * fdim + (k - fdim)];
  else if (k < Kreal) v = grel[(size_t)etype[e] * 100 + (k - fdim2)];
  else                v = 0.f;
  Af[idx] = f2bf(v);
}

// ---------------- dense WMMA GEMM + fused attention logits ----------------
// C[E,200] = A @ W^T + bias; also  eij[e,h] = exp(-leaky_relu(sum_o aatt[h,o]*C[e,h*100+o]))
// One wave per 16-edge tile; 13 N-tiles of accumulators in registers.
// waves_per_eu(2): lift the default occupancy-driven VGPR cap (~128) that was
// spilling the 104 accumulator VGPRs to scratch; ~160 VGPRs live here.
template <int KT>
__global__ __launch_bounds__(32)
__attribute__((amdgpu_waves_per_eu(2)))
void k_gemm(const __bf16* __restrict__ Af, const __bf16* __restrict__ Wf,
            const float* __restrict__ bias, const float* __restrict__ aatt,
            float* __restrict__ C, float* __restrict__ eij) {
  const int lane  = threadIdx.x;
  const int hi    = lane >> 4;
  const int l16   = lane & 15;
  const int etile = blockIdx.x;

  v8f acc[NT_];
#pragma unroll
  for (int nt = 0; nt < NT_; ++nt) acc[nt] = v8f{0.f,0.f,0.f,0.f,0.f,0.f,0.f,0.f};

  const __bf16* ap = Af + ((size_t)etile * KT * 32 + lane) * 16;
  const __bf16* wp = Wf + (size_t)lane * 16;
  for (int kt = 0; kt < KT; ++kt) {
    v16bf a = *(const v16bf*)(ap + (size_t)kt * 512);
    const __bf16* wk = wp + (size_t)kt * (NT_ * 512);
#pragma unroll
    for (int base = 0; base < NT_; base += 4) {
      v16bf bb[4];
#pragma unroll
      for (int q = 0; q < 4; ++q)
        if (base + q < NT_) bb[q] = *(const v16bf*)(wk + (size_t)(base + q) * 512);
#pragma unroll
      for (int q = 0; q < 4; ++q)
        if (base + q < NT_)
          acc[base + q] = __builtin_amdgcn_wmma_f32_16x16x32_bf16(
              false, a, false, bb[q], (short)0, acc[base + q], false, false);
    }
  }

  // Epilogue: add bias, store C, and accumulate per-edge attention-logit dots.
  // D layout: VGPR v -> (edge M = v + 8*hi, output N = nt*16 + l16)
  float pl0[8], pl1[8];
#pragma unroll
  for (int v = 0; v < 8; ++v) { pl0[v] = 0.f; pl1[v] = 0.f; }

#pragma unroll
  for (int nt = 0; nt < NT_; ++nt) {
    int n = nt * 16 + l16;
    if (n < NOUT) {
      float bs = bias[n];
      float wl = aatt[n];     // a_att laid out [h][100] == flat [200]
#pragma unroll
      for (int v = 0; v < 8; ++v) {
        float cv = acc[nt][v] + bs;
        int e = etile * 16 + v + 8 * hi;
        C[(size_t)e * NOUT + n] = cv;
        if (n < OUT_) pl0[v] += wl * cv; else pl1[v] += wl * cv;
      }
    }
  }
  // butterfly reduce across the 16 lanes of each half (wave32; xor<16 stays in half)
#pragma unroll
  for (int m = 1; m < 16; m <<= 1) {
#pragma unroll
    for (int v = 0; v < 8; ++v) {
      pl0[v] += __shfl_xor(pl0[v], m, 32);
      pl1[v] += __shfl_xor(pl1[v], m, 32);
    }
  }
  if (l16 < H_) {
#pragma unroll
    for (int v = 0; v < 8; ++v) {
      float s  = (l16 == 0) ? pl0[v] : pl1[v];
      float lr = s > 0.f ? s : 0.2f * s;        // leaky_relu(0.2)
      int e = etile * 16 + v + 8 * hi;
      eij[(size_t)e * H_ + l16] = __expf(-lr);
    }
  }
}

// ---------------- attention softmax pieces ----------------
__global__ void k_denom(const float* __restrict__ eij, const int* __restrict__ ends,
                        float* __restrict__ denom, int total) {
  int idx = blockIdx.x * blockDim.x + threadIdx.x;
  if (idx >= total) return;
  int e = idx >> 1, h = idx & 1;
  atomicAdd(&denom[(size_t)ends[e] * H_ + h], eij[idx]);
}

__global__ void k_scatter(const float* __restrict__ C, const float* __restrict__ eij,
                          const float* __restrict__ denom, const int* __restrict__ ends,
                          float* __restrict__ hacc) {
  int idx = blockIdx.x * blockDim.x + threadIdx.x;
  if (idx >= E_EDGES * NOUT) return;
  int e = idx / NOUT, o = idx % NOUT;
  int h = o / OUT_;
  int dst = ends[e];
  float alpha = eij[(size_t)e * H_ + h] / denom[(size_t)dst * H_ + h];
  atomicAdd(&hacc[(size_t)dst * NOUT + o], alpha * C[(size_t)e * NOUT + o]);
}

// elu(segsum), sigmoid gate, combine, elu, per-head L2 normalize
__global__ void k_combine(const float* __restrict__ hina, const float* __restrict__ houta,
                          const float* __restrict__ Wa, const float* __restrict__ ba,
                          float* __restrict__ hf, int n) {
  int idx = blockIdx.x * blockDim.x + threadIdx.x;
  if (idx >= n * H_) return;
  int i = idx >> 1, h = idx & 1;
  const float* pi = hina  + (size_t)i * NOUT + h * OUT_;
  const float* po = houta + (size_t)i * NOUT + h * OUT_;
  float s = ba[0];
  for (int o = 0; o < OUT_; ++o) s += Wa[o] * eluf(pi[o]) + Wa[OUT_ + o] * eluf(po[o]);
  float al = 1.f / (1.f + __expf(-s));
  float ss = 0.f;
  for (int o = 0; o < OUT_; ++o) {
    float v = al * eluf(pi[o]) + (1.f - al) * eluf(po[o]);
    float t = eluf(v);
    ss += t * t;
  }
  float inv = 1.f / fmaxf(sqrtf(ss), 1e-12f);
  float* op = hf + (size_t)i * NOUT + h * OUT_;
  for (int o = 0; o < OUT_; ++o) {
    float v = al * eluf(pi[o]) + (1.f - al) * eluf(po[o]);
    op[o] = eluf(v) * inv;
  }
}

// entity residual + per-head L2 normalize; one block per node
__global__ void k_final(const float* __restrict__ xn, const float* __restrict__ Went,
                        const float* __restrict__ bent, const float* __restrict__ h2,
                        float* __restrict__ out) {
  __shared__ float vbuf[NOUT];
  __shared__ float norms[H_];
  int i = blockIdx.x, t = threadIdx.x;
  if (t < NOUT) {
    const float* xr = xn + (size_t)i * XS_;
    const float* wr = Went + (size_t)t * XS_;
    float s = bent[t];
    for (int k = 0; k < XS_; ++k) s += xr[k] * wr[k];
    vbuf[t] = s + h2[(size_t)i * NOUT + t];
  }
  __syncthreads();
  if (t < H_) {
    float ss = 0.f;
    for (int o = 0; o < OUT_; ++o) { float v = vbuf[t * OUT_ + o]; ss += v * v; }
    norms[t] = fmaxf(sqrtf(ss), 1e-12f);
  }
  __syncthreads();
  if (t < NOUT) out[(size_t)i * NOUT + t] = vbuf[t] / norms[t / OUT_];
}

__global__ void k_gprime(const float* __restrict__ g, const float* __restrict__ Wrel,
                         const float* __restrict__ brel, float* __restrict__ out) {
  int idx = blockIdx.x * blockDim.x + threadIdx.x;
  if (idx >= M_REL * NOUT) return;
  int m = idx / NOUT, o = idx % NOUT;
  const float* gr = g + (size_t)m * 100;
  const float* wr = Wrel + (size_t)o * 100;
  float s = brel[o];
  for (int k = 0; k < 100; ++k) s += gr[k] * wr[k];
  out[idx] = s;
}

// ---------------- host launch ----------------
extern "C" void kernel_launch(void* const* d_in, const int* in_sizes, int n_in,
                              void* d_out, int out_size, void* d_ws, size_t ws_size,
                              hipStream_t stream) {
  (void)in_sizes; (void)n_in; (void)out_size; (void)ws_size;
  const float* x     = (const float*)d_in[0];
  const float* g     = (const float*)d_in[1];
  const float* W1in  = (const float*)d_in[2];
  const float* b1in  = (const float*)d_in[3];
  const float* a1in  = (const float*)d_in[4];
  const float* W1out = (const float*)d_in[5];
  const float* b1out = (const float*)d_in[6];
  const float* a1out = (const float*)d_in[7];
  const float* Wa1   = (const float*)d_in[8];
  const float* ba1   = (const float*)d_in[9];
  const float* W2in  = (const float*)d_in[10];
  const float* b2in  = (const float*)d_in[11];
  const float* a2in  = (const float*)d_in[12];
  const float* W2out = (const float*)d_in[13];
  const float* b2out = (const float*)d_in[14];
  const float* a2out = (const float*)d_in[15];
  const float* Wa2   = (const float*)d_in[16];
  const float* ba2   = (const float*)d_in[17];
  const float* Went  = (const float*)d_in[18];
  const float* bent  = (const float*)d_in[19];
  const float* Wrel  = (const float*)d_in[20];
  const float* brel  = (const float*)d_in[21];
  const int* edge_idx = (const int*)d_in[22];
  const int* etype    = (const int*)d_in[23];
  const int* row = edge_idx;
  const int* col = edge_idx + E_EDGES;

  char* ws = (char*)d_ws;
  size_t off = 0;
  auto walloc = [&](size_t bytes) -> void* {
    void* p = ws + off;
    off += (bytes + 255) & ~(size_t)255;
    return p;
  };
  float*  xn    = (float*)walloc((size_t)N_NODES * XS_ * 4);
  __bf16* Wf1i  = (__bf16*)walloc((size_t)NPAD * K1P * 2);
  __bf16* Wf1o  = (__bf16*)walloc((size_t)NPAD * K1P * 2);
  __bf16* Wf2i  = (__bf16*)walloc((size_t)NPAD * K2P * 2);
  __bf16* Wf2o  = (__bf16*)walloc((size_t)NPAD * K2P * 2);
  __bf16* Afrag = (__bf16*)walloc((size_t)E_EDGES * K2P * 2);  // reused by both layers
  float*  cbuf  = (float*)walloc((size_t)E_EDGES * NOUT * 4);  // reused x4
  float*  eij   = (float*)walloc((size_t)E_EDGES * H_ * 4);
  float*  denom = (float*)walloc((size_t)N_NODES * H_ * 4);
  float*  hinb  = (float*)walloc((size_t)N_NODES * NOUT * 4);
  float*  houtb = (float*)walloc((size_t)N_NODES * NOUT * 4);
  float*  h1    = (float*)walloc((size_t)N_NODES * NOUT * 4);
  float*  h2    = (float*)walloc((size_t)N_NODES * NOUT * 4);

  // prep
  k_l2norm_rows<<<(N_NODES + 255) / 256, 256, 0, stream>>>(x, xn, N_NODES);
  k_prep_wfrag<<<(NPAD * K1P + 255) / 256, 256, 0, stream>>>(W1in,  Wf1i, K1, K1P / 32);
  k_prep_wfrag<<<(NPAD * K1P + 255) / 256, 256, 0, stream>>>(W1out, Wf1o, K1, K1P / 32);
  k_prep_wfrag<<<(NPAD * K2P + 255) / 256, 256, 0, stream>>>(W2in,  Wf2i, K2, K2P / 32);
  k_prep_wfrag<<<(NPAD * K2P + 255) / 256, 256, 0, stream>>>(W2out, Wf2o, K2, K2P / 32);

  auto run_rest = [&](const int* ends, float* hacc) {
    int nl = E_EDGES * H_;
    hipMemsetAsync(denom, 0, (size_t)N_NODES * H_ * 4, stream);
    k_denom<<<(nl + 255) / 256, 256, 0, stream>>>(eij, ends, denom, nl);
    hipMemsetAsync(hacc, 0, (size_t)N_NODES * NOUT * 4, stream);
    int tot = E_EDGES * NOUT;
    k_scatter<<<(tot + 255) / 256, 256, 0, stream>>>(cbuf, eij, denom, ends, hacc);
  };

  // ---- layer 1: gather h_ijk once (shared by in/out directions) ----
  {
    int tot = E_EDGES * K1P;
    k_gather_A<<<(tot + 255) / 256, 256, 0, stream>>>(xn, XS_, g, row, col, etype,
                                                      Afrag, K1, K1P / 32);
  }
  k_gemm<K1P / 32><<<ETILES, 32, 0, stream>>>(Afrag, Wf1i, b1in, a1in, cbuf, eij);
  run_rest(col, hinb);                                   // inbound: ends = col
  k_gemm<K1P / 32><<<ETILES, 32, 0, stream>>>(Afrag, Wf1o, b1out, a1out, cbuf, eij);
  run_rest(row, houtb);                                  // outbound: ends = row
  k_combine<<<(N_NODES * H_ + 255) / 256, 256, 0, stream>>>(hinb, houtb, Wa1, ba1, h1, N_NODES);

  // ---- layer 2 ----
  {
    int tot = E_EDGES * K2P;
    k_gather_A<<<(tot + 255) / 256, 256, 0, stream>>>(h1, NOUT, g, row, col, etype,
                                                      Afrag, K2, K2P / 32);
  }
  k_gemm<K2P / 32><<<ETILES, 32, 0, stream>>>(Afrag, Wf2i, b2in, a2in, cbuf, eij);
  run_rest(col, hinb);
  k_gemm<K2P / 32><<<ETILES, 32, 0, stream>>>(Afrag, Wf2o, b2out, a2out, cbuf, eij);
  run_rest(row, houtb);
  k_combine<<<(N_NODES * H_ + 255) / 256, 256, 0, stream>>>(hinb, houtb, Wa2, ba2, h2, N_NODES);

  // outputs: h_prime [N,200] then g_prime [M,200]
  float* out_h = (float*)d_out;
  float* out_g = out_h + (size_t)N_NODES * NOUT;
  k_final<<<N_NODES, 256, 0, stream>>>(xn, Went, bent, h2, out_h);
  k_gprime<<<(M_REL * NOUT + 255) / 256, 256, 0, stream>>>(g, Wrel, brel, out_g);
}
